// StandardRNN_63651415327171
// MI455X (gfx1250) — compile-verified, matrix operand order checked by reading, not run on hardware
//
#include <hip/hip_runtime.h>
#include <hip/hip_bf16.h>
#include <math.h>

// ---------------------------------------------------------------------------
// Types for CDNA5 WMMA (wave32): v16h A/B fragments, v8f accumulators.
// ---------------------------------------------------------------------------
typedef _Float16 half8  __attribute__((ext_vector_type(8)));
typedef _Float16 half16 __attribute__((ext_vector_type(16)));
typedef float    float8 __attribute__((ext_vector_type(8)));
typedef unsigned int u32x4 __attribute__((ext_vector_type(4)));
typedef int          i32x4 __attribute__((ext_vector_type(4)));
typedef int          i32x8 __attribute__((ext_vector_type(8)));

#define B_DIM   4096
#define SEQ     128
#define EV      64
#define RNN_IN  256
#define H_DIM   256
#define T_STEPS 32
#define LABELS  100
#define LAB_PAD 112   // 7 * 16

#if defined(__has_builtin)
#if __has_builtin(__builtin_amdgcn_tensor_load_to_lds) && \
    __has_builtin(__builtin_amdgcn_s_wait_tensorcnt)
#define HAVE_TDM 1
#endif
#endif
#ifndef HAVE_TDM
#define HAVE_TDM 0
#endif

// ---------------------------------------------------------------------------
// Build a 16-half fragment from two contiguous 8-half (16B) chunks:
// CDNA5 16-bit A/B layout: per lane, elements 0..7 = K[base..base+7],
// elements 8..15 = K[base+16..base+23]; base = k0 + (lane>=16 ? 8 : 0).
// Works for both LDS (ds_load_b128) and global (global_load_b128) pointers.
// ---------------------------------------------------------------------------
__device__ __forceinline__ half16 make_frag(const _Float16* p) {
  half8 lo = *(const half8*)(p);
  half8 hi = *(const half8*)(p + 16);
  half16 r;
#pragma unroll
  for (int i = 0; i < 8; ++i) { r[i] = lo[i]; r[i + 8] = hi[i]; }
  return r;
}

__device__ __forceinline__ float8 wmma16(half16 a, half16 b, float8 c) {
  // v_wmma_f32_16x16x32_f16  (8-arg form, probe-confirmed)
  return __builtin_amdgcn_wmma_f32_16x16x32_f16(
      /*neg_a=*/false, a, /*neg_b=*/false, b,
      /*c_mod=*/(short)0, c, /*reuse_a=*/false, /*reuse_b=*/false);
}

__device__ __forceinline__ float sigmoidf_fast(float x) {
  return 1.0f / (1.0f + __expf(-x));
}

#if HAVE_TDM
// ---------------------------------------------------------------------------
// Issue a TDM 2D-tile load: [32 rows x 256 fp16], row stride 8192 elements
// (16KB), global -> LDS. Descriptor per CDNA5 ISA.
// ---------------------------------------------------------------------------
__device__ __forceinline__ void tdm_load_x(const _Float16* gsrc,
                                           _Float16* lds_dst) {
  const unsigned long long ga = (unsigned long long)(uintptr_t)gsrc;
  // LDS aperture: flat addr[31:0] is the LDS byte offset.
  const unsigned int lds_off = (unsigned int)(uintptr_t)lds_dst;
  // D# group0: count=1 | lds_addr | global_addr[56:0] | type=2.
  u32x4 g0 = {1u, lds_off, (unsigned int)ga,
              ((unsigned int)(ga >> 32) & 0x01FFFFFFu) | 0x80000000u};
  // D# group1: data_size=2B; tensor_dim0=8192, tensor_dim1=32;
  //            tile_dim0=256, tile_dim1=32; tensor_dim0_stride=8192.
  i32x8 g1 = {0x00010000, 0x20000000, 0x00200000, 0x01000000,
              32,         8192,       0,          0};
  i32x4 z4 = {0, 0, 0, 0};
#if __clang_major__ >= 23
  i32x8 z8 = {0, 0, 0, 0, 0, 0, 0, 0};
  __builtin_amdgcn_tensor_load_to_lds(g0, g1, z4, z4, z8, 0);
#else
  __builtin_amdgcn_tensor_load_to_lds(g0, g1, z4, z4, 0);
#endif
}
#endif

// ---------------------------------------------------------------------------
// fp32 -> fp16 repack kernels (weights stay resident in L2 afterwards)
// ---------------------------------------------------------------------------
__global__ void f32_to_f16_kernel(const float* __restrict__ src,
                                  _Float16* __restrict__ dst, int n) {
  int i = blockIdx.x * blockDim.x + threadIdx.x;
  if (i < n) dst[i] = (_Float16)src[i];
}

__global__ void conv_outw_kernel(const float* __restrict__ src,
                                 _Float16* __restrict__ dst) {
  int i = blockIdx.x * blockDim.x + threadIdx.x;  // over LAB_PAD*H_DIM
  if (i < LAB_PAD * H_DIM) {
    int r = i >> 8, c = i & 255;
    dst[i] = (r < LABELS) ? (_Float16)src[r * H_DIM + c] : (_Float16)0.0f;
  }
}

// ---------------------------------------------------------------------------
// Embedding gather + FC:  evs[r, 0:64] = concat(len_emb[len_x[r]],
//                                               ipd_emb[ipd_x[r]]) @ fc_w^T + b
// A (16x128, gathered fp32->fp16 on the fly), B = fc_w rows. 4 K-steps of
// wmma per 16x16 tile, 4 column tiles. One wave per 16-row tile.
// ---------------------------------------------------------------------------
__global__ __launch_bounds__(256) void embed_fc_kernel(
    const int* __restrict__ len_x, const int* __restrict__ ipd_x,
    const float* __restrict__ len_emb, const float* __restrict__ ipd_emb,
    const _Float16* __restrict__ fc_w_h, const float* __restrict__ fc_b,
    _Float16* __restrict__ rnn_in_h) {
  const int wave = threadIdx.x >> 5;
  const int lane = threadIdx.x & 31;
  const int m    = lane & 15;
  const int hi8  = (lane >> 4) * 8;          // 0 or 8
  const int r0   = (blockIdx.x * 8 + wave) * 16;
  const int r    = r0 + m;

  const int li = len_x[r];
  const int pi = ipd_x[r];

  // Gather A fragments (K tiles 0,32 from len_emb; 64,96 from ipd_emb).
  half16 afrag[4];
#pragma unroll
  for (int kt = 0; kt < 4; ++kt) {
    const float* tab = (kt < 2) ? (len_emb + (size_t)li * 64)
                                : (ipd_emb + (size_t)pi * 64);
    const int kbase = (kt & 1) * 32 + hi8;
    half16 f;
#pragma unroll
    for (int e = 0; e < 8; ++e) {
      f[e]     = (_Float16)tab[kbase + e];
      f[e + 8] = (_Float16)tab[kbase + 16 + e];
    }
    afrag[kt] = f;
  }

#pragma unroll
  for (int nt = 0; nt < 4; ++nt) {
    const int n0 = nt * 16;
    float8 acc = {};
#pragma unroll
    for (int kt = 0; kt < 4; ++kt) {
      half16 b = make_frag(fc_w_h + (size_t)(n0 + m) * 128 + kt * 32 + hi8);
      acc = wmma16(afrag[kt], b, acc);
    }
    const int col  = n0 + m;
    const float bv = fc_b[col];
#pragma unroll
    for (int j = 0; j < 8; ++j) {
      const int row = r0 + j + hi8;
      rnn_in_h[(size_t)row * EV + col] = (_Float16)(acc[j] + bv);
    }
  }
}

// ---------------------------------------------------------------------------
// GRU: 32 sequential steps. Each block owns 32 batch rows; h double-buffered
// in LDS (fp16). x-tiles are DMA'd by the Tensor Data Mover into a double
// buffer, software-pipelined one step ahead: at step t, wave 0 waits on
// TENSORcnt for x_t (already in flight since step t-1), then immediately
// issues the DMA for x_{t+1} so the fetch overlaps all of step t's WMMAs.
// A wave computes a 16x16 tile of H-columns: acc_r, acc_z, gi_n, gh_n all in
// registers (48 WMMAs per tile), then sigmoid/tanh/update.
// ---------------------------------------------------------------------------
__global__ __launch_bounds__(256) void gru_kernel(
    const _Float16* __restrict__ xall,   // [B, 32, 256] fp16
    const _Float16* __restrict__ w_ih,   // [768, 256] fp16
    const _Float16* __restrict__ w_hh,   // [768, 256] fp16
    const float* __restrict__ b_ih, const float* __restrict__ b_hh,
    _Float16* __restrict__ h_out) {      // [B, 256] fp16
  __shared__ _Float16 xs[2][32][RNN_IN];     // 32 KB (double buffer)
  __shared__ _Float16 hs[2][32][H_DIM];      // 32 KB (double buffer)

  const int tid  = threadIdx.x;
  const int wave = tid >> 5;
  const int lane = tid & 31;
  const int m    = lane & 15;
  const int hi8  = (lane >> 4) * 8;
  const int b0   = blockIdx.x * 32;
  const _Float16* xbase = xall + (size_t)b0 * (T_STEPS * RNN_IN);

  // h0 = 0
  for (int i = tid; i < 32 * H_DIM; i += 256) hs[0][0][i] = (_Float16)0.0f;

#if HAVE_TDM
  // Prime the pipeline: DMA x_0 into buffer 0.
  if (wave == 0) tdm_load_x(xbase, &xs[0][0][0]);
#endif
  __syncthreads();

  int cur = 0;
  for (int t = 0; t < T_STEPS; ++t) {
    const int xb = t & 1;
#if HAVE_TDM
    // x_t was issued one step ago; drain TENSORcnt, publish via barrier,
    // then immediately put x_{t+1} in flight (its buffer was last read at
    // step t-1, so the barrier proves it is free).
    if (wave == 0) __builtin_amdgcn_s_wait_tensorcnt(0);
    __syncthreads();
    if (wave == 0 && t + 1 < T_STEPS)
      tdm_load_x(xbase + (size_t)(t + 1) * RNN_IN, &xs[xb ^ 1][0][0]);
#else
    // Fallback: cooperative vectorized staging.
    for (int i = tid; i < (32 * RNN_IN) / 8; i += 256) {
      const int row = i >> 5;
      const int kc  = (i & 31) * 8;
      *(half8*)&xs[xb][row][kc] =
          *(const half8*)(xbase + (size_t)row * (T_STEPS * RNN_IN) +
                          (size_t)t * RNN_IN + kc);
    }
    __syncthreads();
#endif

    const int nxt = cur ^ 1;
    // 2 row-tiles x 16 col-tiles = 32 tasks; 8 waves -> 4 tasks each.
    for (int task = wave; task < 32; task += 8) {
      const int r0  = (task >> 4) * 16;
      const int c0  = (task & 15) * 16;
      const int col = c0 + m;

      // Hint the n-gate rows into cache ahead of use.
      __builtin_prefetch(w_ih + (size_t)(512 + col) * RNN_IN, 0, 1);
      __builtin_prefetch(w_hh + (size_t)(512 + col) * RNN_IN, 0, 1);

      float8 accr = {}, accz = {}, accin = {}, acchn = {};
#pragma unroll
      for (int kk = 0; kk < 8; ++kk) {
        const int ko = kk * 32 + hi8;
        half16 ax = make_frag(&xs[xb][r0 + m][ko]);
        half16 ah = make_frag(&hs[cur][r0 + m][ko]);
        half16 bri = make_frag(w_ih + (size_t)(col)       * RNN_IN + ko);
        half16 bzi = make_frag(w_ih + (size_t)(256 + col) * RNN_IN + ko);
        half16 bni = make_frag(w_ih + (size_t)(512 + col) * RNN_IN + ko);
        half16 brh = make_frag(w_hh + (size_t)(col)       * RNN_IN + ko);
        half16 bzh = make_frag(w_hh + (size_t)(256 + col) * RNN_IN + ko);
        half16 bnh = make_frag(w_hh + (size_t)(512 + col) * RNN_IN + ko);
        accr  = wmma16(ax, bri, accr);
        accr  = wmma16(ah, brh, accr);
        accz  = wmma16(ax, bzi, accz);
        accz  = wmma16(ah, bzh, accz);
        accin = wmma16(ax, bni, accin);
        acchn = wmma16(ah, bnh, acchn);
      }

      const float br  = b_ih[col]       + b_hh[col];
      const float bz  = b_ih[256 + col] + b_hh[256 + col];
      const float bin = b_ih[512 + col];
      const float bhn = b_hh[512 + col];
#pragma unroll
      for (int j = 0; j < 8; ++j) {
        const int row = r0 + j + hi8;
        const float rg = sigmoidf_fast(accr[j] + br);
        const float zg = sigmoidf_fast(accz[j] + bz);
        const float ng = tanhf(accin[j] + bin + rg * (acchn[j] + bhn));
        const float ho = (float)hs[cur][row][col];
        hs[nxt][row][col] = (_Float16)((1.0f - zg) * ng + zg * ho);
      }
    }
    cur ^= 1;
    __syncthreads();
  }

  // Emit final hidden state (fp16) for the output head.
  for (int i = tid; i < (32 * H_DIM) / 8; i += 256) {
    const int row = i >> 5;
    const int kc  = (i & 31) * 8;
    *(half8*)(h_out + (size_t)(b0 + row) * H_DIM + kc) =
        *(const half8*)&hs[cur][row][kc];
  }
}

// ---------------------------------------------------------------------------
// Output head: logits = h @ out_w^T + out_b  ([4096,256] x [256,112pad])
// ---------------------------------------------------------------------------
__global__ __launch_bounds__(256) void out_head_kernel(
    const _Float16* __restrict__ h16,      // [B, 256]
    const _Float16* __restrict__ out_w_h,  // [112, 256] (zero padded)
    const float* __restrict__ out_b, float* __restrict__ logits) {
  const int wave = threadIdx.x >> 5;
  const int lane = threadIdx.x & 31;
  const int m    = lane & 15;
  const int hi8  = (lane >> 4) * 8;

  const int task  = blockIdx.x * 8 + wave;  // 256 row-tiles * 7 col-tiles
  const int rtile = task / 7;
  const int ct    = task % 7;
  const int r0    = rtile * 16;
  const int col   = ct * 16 + m;

  float8 acc = {};
#pragma unroll
  for (int kk = 0; kk < 8; ++kk) {
    const int ko = kk * 32 + hi8;
    half16 a = make_frag(h16 + (size_t)(r0 + m) * H_DIM + ko);
    half16 b = make_frag(out_w_h + (size_t)col * H_DIM + ko);
    acc = wmma16(a, b, acc);
  }
  if (col < LABELS) {
    const float bv = out_b[col];
#pragma unroll
    for (int j = 0; j < 8; ++j)
      logits[(size_t)(r0 + j + hi8) * LABELS + col] = acc[j] + bv;
  }
}

// ---------------------------------------------------------------------------
// Host-side launcher.
// Inputs (setup_inputs order):
//  0 len_x [B,SEQ] i32        1 ipd_x [B,SEQ] i32
//  2 len_emb [1500,64] f32    3 ipd_emb [256,64] f32
//  4 fc_w [64,128] f32        5 fc_b [64] f32
//  6 w_ih [768,256] f32       7 w_hh [768,256] f32
//  8 b_ih [768] f32           9 b_hh [768] f32
// 10 out_w [100,256] f32     11 out_b [100] f32
// ---------------------------------------------------------------------------
extern "C" void kernel_launch(void* const* d_in, const int* in_sizes, int n_in,
                              void* d_out, int out_size, void* d_ws,
                              size_t ws_size, hipStream_t stream) {
  const int*   len_x   = (const int*)d_in[0];
  const int*   ipd_x   = (const int*)d_in[1];
  const float* len_emb = (const float*)d_in[2];
  const float* ipd_emb = (const float*)d_in[3];
  const float* fc_w    = (const float*)d_in[4];
  const float* fc_b    = (const float*)d_in[5];
  const float* w_ih    = (const float*)d_in[6];
  const float* w_hh    = (const float*)d_in[7];
  const float* b_ih    = (const float*)d_in[8];
  const float* b_hh    = (const float*)d_in[9];
  const float* out_w   = (const float*)d_in[10];
  const float* out_b   = (const float*)d_in[11];
  float* logits = (float*)d_out;

  // Workspace layout (bytes, 256-aligned blocks).
  char* ws = (char*)d_ws;
  _Float16* fc_w_h   = (_Float16*)(ws + 0);           //  16,384 B
  _Float16* w_ih_h   = (_Float16*)(ws + 16384);       // 393,216 B
  _Float16* w_hh_h   = (_Float16*)(ws + 409600);      // 393,216 B
  _Float16* out_w_h  = (_Float16*)(ws + 802816);      //  57,344 B
  _Float16* rnn_in_h = (_Float16*)(ws + 860160);      //  64 MB
  _Float16* h16      = (_Float16*)(ws + 860160 + (size_t)B_DIM * SEQ * EV * 2);

  const int thr = 256;
  // Weight repack fp32 -> fp16.
  f32_to_f16_kernel<<<(EV * 128 + thr - 1) / thr, thr, 0, stream>>>(
      fc_w, fc_w_h, EV * 128);
  f32_to_f16_kernel<<<(768 * 256 + thr - 1) / thr, thr, 0, stream>>>(
      w_ih, w_ih_h, 768 * 256);
  f32_to_f16_kernel<<<(768 * 256 + thr - 1) / thr, thr, 0, stream>>>(
      w_hh, w_hh_h, 768 * 256);
  conv_outw_kernel<<<(LAB_PAD * H_DIM + thr - 1) / thr, thr, 0, stream>>>(
      out_w, out_w_h);

  // Embedding gather + FC -> rnn_in (fp16).
  embed_fc_kernel<<<(B_DIM * SEQ / 16) / 8, thr, 0, stream>>>(
      len_x, ipd_x, len_emb, ipd_emb, fc_w_h, fc_b, rnn_in_h);

  // Recurrent GRU scan.
  gru_kernel<<<B_DIM / 32, thr, 0, stream>>>(rnn_in_h, w_ih_h, w_hh_h, b_ih,
                                             b_hh, h16);

  // Output head.
  out_head_kernel<<<(256 * 7) / 8, thr, 0, stream>>>(h16, out_w_h, out_b,
                                                     logits);
}
